// ParallelMPNNModel_66125316489911
// MI455X (gfx1250) — matrix operand order repaired and checked by reading, not compiled, not run on hardware
//
#include <hip/hip_runtime.h>
#include <hip/hip_bf16.h>

// ---------------- problem constants ----------------
#define GG   16      // graphs
#define NN   96      // nodes per graph
#define HH   128     // hidden
#define EE   24576   // edges
#define LL   3       // layers

typedef __attribute__((ext_vector_type(16))) _Float16 v16h;
typedef __attribute__((ext_vector_type(8)))  float    v8f;

// ---------------- utility kernels ----------------

__global__ void zero_kernel(uint4* p, int n16) {
    uint4 z = {0u, 0u, 0u, 0u};
    for (int i = blockIdx.x * blockDim.x + threadIdx.x; i < n16;
         i += gridDim.x * blockDim.x)
        p[i] = z;
}

// node embeddings: nf[n,h] = sum_c atom_tables[c, x[n,c], h]
__global__ __launch_bounds__(128) void node_embed_kernel(
    const int* __restrict__ x, const float* __restrict__ atom_tables,
    float* __restrict__ nf) {
    int n = blockIdx.x, h = threadIdx.x;
    float v = 0.f;
#pragma unroll
    for (int c = 0; c < 9; ++c)
        v += atom_tables[((size_t)c * 119 + x[n * 9 + c]) * HH + h];
    nf[(size_t)n * HH + h] = v;
}

// edge embeddings scattered into dense f16 tensor Ebuf[g, src, dst, h]
__global__ __launch_bounds__(128) void edge_scatter_kernel(
    const int* __restrict__ edge_attr, const int* __restrict__ esrc,
    const int* __restrict__ edst, const float* __restrict__ bond_tables,
    _Float16* __restrict__ Ebuf) {
    int e = blockIdx.x, h = threadIdx.x;
    float v = 0.f;
#pragma unroll
    for (int c = 0; c < 3; ++c)
        v += bond_tables[((size_t)c * 6 + edge_attr[e * 3 + c]) * HH + h];
    int s = esrc[e], d = edst[e];
    int g = s / NN, li = s % NN, lj = d % NN;
    Ebuf[(((size_t)g * NN + li) * NN + lj) * HH + h] = (_Float16)v;
}

// per-layer node linears for both streams:
//   C1 = z@W_m1 + b_m1 + b_me + b_mg   (graph_fts is always zero -> msgg = b_mg)
//   M2 = z@W_m2 + b_m2
//   ZO = z@W_o1 + b_o1
__global__ __launch_bounds__(128) void node_linear_kernel(
    const float* __restrict__ nf, const float* __restrict__ hid,
    const float* __restrict__ W_m1, const float* __restrict__ b_m1,
    const float* __restrict__ W_m2, const float* __restrict__ b_m2,
    const float* __restrict__ W_o1, const float* __restrict__ b_o1,
    const float* __restrict__ b_me, const float* __restrict__ b_mg,
    float* __restrict__ C1a, float* __restrict__ C1b,
    float* __restrict__ M2a, float* __restrict__ M2b,
    float* __restrict__ ZOa, float* __restrict__ ZOb, int layer) {
    __shared__ float z[2 * HH];
    int node = blockIdx.x, h = threadIdx.x;
    z[h]      = nf[(size_t)node * HH + h];
    z[HH + h] = hid[(size_t)node * HH + h];
    __syncthreads();
    for (int s = 0; s < 2; ++s) {
        const float* w1 = W_m1 + (size_t)(layer * 2 + s) * 2 * HH * HH;
        const float* w2 = W_m2 + (size_t)(layer * 2 + s) * 2 * HH * HH;
        const float* wo = W_o1 + (size_t)(layer * 2 + s) * 2 * HH * HH;
        float a1 = 0.f, a2 = 0.f, a3 = 0.f;
        for (int k = 0; k < 2 * HH; ++k) {
            float zk = z[k];
            a1 += zk * w1[(size_t)k * HH + h];
            a2 += zk * w2[(size_t)k * HH + h];
            a3 += zk * wo[(size_t)k * HH + h];
        }
        int bo = (layer * 2 + s) * HH + h;
        size_t o = (size_t)node * HH + h;
        (s ? C1b : C1a)[o] = a1 + b_m1[bo] + b_me[bo] + b_mg[bo];
        (s ? M2b : M2a)[o] = a2 + b_m2[bo];
        (s ? ZOb : ZOa)[o] = a3 + b_o1[bo];
    }
}

// ---------------- fused WMMA message kernel ----------------
// One workgroup per (g, j=dst). 8 waves; wave w owns N-columns [16w,16w+16).
// Pipeline: T = E[g,:,j,:] ; D1 = relu(T@Wme + C1[g,j,:] + M2[g,i,:])
//           D2 = relu(D1@Wmlp1 + b1) ; D3 = D2@Wmlp2 + b2 ; out = max_i D3
// E tile staged with GLOBAL_LOAD_ASYNC_TO_LDS_B128 (ASYNCcnt path, no VGPR
// round-trip). Weights staged transposed (Wt[n][k]) via VALU f32->f16.

__device__ __forceinline__ void gemm_96x128x128(
    const _Float16* __restrict__ Tb,   // [96][128] LDS, A operand
    const _Float16* __restrict__ Wt,   // [128][128] LDS, transposed weight
    int lane, int ncol, v8f acc[6]) {
    const int ah = (lane & 16) ? 8  : 0;   // A K-chunk select (doc layout)
    const int bh = (lane & 16) ? 8  : 0;   // B K-chunk select (doc layout)
    const int ml = lane & 15;
#pragma unroll
    for (int kb = 0; kb < HH; kb += 32) {
        v16h bf;
        const _Float16* bp = &Wt[ncol * HH + kb + bh];
#pragma unroll
        for (int e = 0; e < 8; ++e) { bf[e] = bp[e]; bf[8 + e] = bp[16 + e]; }
#pragma unroll
        for (int mt = 0; mt < 6; ++mt) {
            const _Float16* ap = &Tb[(mt * 16 + ml) * HH + kb + ah];
            v16h af;
#pragma unroll
            for (int e = 0; e < 8; ++e) { af[e] = ap[e]; af[8 + e] = ap[16 + e]; }
            acc[mt] = __builtin_amdgcn_wmma_f32_16x16x32_f16(
                false, af, false, bf, (short)0, acc[mt], false, false);
        }
    }
}

__global__ __launch_bounds__(256) void fused_msg_kernel(
    const _Float16* __restrict__ Ebuf, const float* __restrict__ C1,
    const float* __restrict__ M2, const float* __restrict__ Wme,
    const float* __restrict__ Wmlp1, const float* __restrict__ Wmlp2,
    const float* __restrict__ bmlp1, const float* __restrict__ bmlp2,
    float* __restrict__ MMAX) {
    __shared__ _Float16 Tb[NN * HH];       // 24 KB tile (ping-ponged)
    __shared__ _Float16 Wt[HH * HH];       // 32 KB transposed weight
    __shared__ float    c1s[HH];           // per-dst bias vector

    const int tid = threadIdx.x;
    const int g = blockIdx.x / NN, j = blockIdx.x % NN;

    // L2 prefetch for the stage-2/3 weights (used later in this block)
    __builtin_prefetch(Wmlp1 + (size_t)tid * 64, 0, 0);
    __builtin_prefetch(Wmlp2 + (size_t)tid * 64, 0, 0);

    // ---- stage in E tile via async global->LDS (96 rows x 128 f16) ----
    // 1536 x 16-byte chunks, 6 per thread, all lanes active (EXEC all-1s).
#pragma unroll
    for (int it = 0; it < 6; ++it) {
        int idx = tid + it * 256;
        int r = idx >> 4, c = idx & 15;
        unsigned long long ga = (unsigned long long)(uintptr_t)(
            Ebuf + (((size_t)g * NN + r) * NN + j) * HH + (size_t)c * 8);
        unsigned int la =
            (unsigned int)(uintptr_t)((char*)Tb + (size_t)idx * 16);
        asm volatile("global_load_async_to_lds_b128 %0, %1, off"
                     :: "v"(la), "v"(ga)
                     : "memory");
    }
    // ---- stage Wme transposed + c1 (VALU path, runs under async copies) ----
    for (int idx = tid; idx < HH * HH; idx += 256) {
        int n = idx >> 7, k = idx & 127;
        Wt[idx] = (_Float16)Wme[(size_t)k * HH + n];
    }
    if (tid < HH) c1s[tid] = C1[((size_t)g * NN + j) * HH + tid];
    asm volatile("s_wait_asynccnt 0x0" ::: "memory");
    __syncthreads();

    const int lane = tid & 31, wv = tid >> 5;
    const int ncol = wv * 16 + (lane & 15);
    const int hh8  = (lane & 16) ? 8 : 0;  // D row offset within tile
    float res[6][8];

    // ---- stage 1: relu(E@Wme + c1 + M2[i]) ----
    {
        v8f acc[6];
#pragma unroll
        for (int t = 0; t < 6; ++t)
#pragma unroll
            for (int e = 0; e < 8; ++e) acc[t][e] = 0.f;
        gemm_96x128x128(Tb, Wt, lane, ncol, acc);
        float c1v = c1s[ncol];
#pragma unroll
        for (int mt = 0; mt < 6; ++mt)
#pragma unroll
            for (int v = 0; v < 8; ++v) {
                int m = mt * 16 + v + hh8;
                float b = c1v + M2[((size_t)g * NN + m) * HH + ncol];
                res[mt][v] = fmaxf(acc[mt][v] + b, 0.f);
            }
    }
    __syncthreads();
#pragma unroll
    for (int mt = 0; mt < 6; ++mt)
#pragma unroll
        for (int v = 0; v < 8; ++v)
            Tb[(mt * 16 + v + hh8) * HH + ncol] = (_Float16)res[mt][v];
    for (int idx = tid; idx < HH * HH; idx += 256) {
        int n = idx >> 7, k = idx & 127;
        Wt[idx] = (_Float16)Wmlp1[(size_t)k * HH + n];
    }
    __syncthreads();

    // ---- stage 2: relu(T1@Wmlp1 + b1) ----
    {
        v8f acc[6];
#pragma unroll
        for (int t = 0; t < 6; ++t)
#pragma unroll
            for (int e = 0; e < 8; ++e) acc[t][e] = 0.f;
        gemm_96x128x128(Tb, Wt, lane, ncol, acc);
        float b1 = bmlp1[ncol];
#pragma unroll
        for (int mt = 0; mt < 6; ++mt)
#pragma unroll
            for (int v = 0; v < 8; ++v)
                res[mt][v] = fmaxf(acc[mt][v] + b1, 0.f);
    }
    __syncthreads();
#pragma unroll
    for (int mt = 0; mt < 6; ++mt)
#pragma unroll
        for (int v = 0; v < 8; ++v)
            Tb[(mt * 16 + v + hh8) * HH + ncol] = (_Float16)res[mt][v];
    for (int idx = tid; idx < HH * HH; idx += 256) {
        int n = idx >> 7, k = idx & 127;
        Wt[idx] = (_Float16)Wmlp2[(size_t)k * HH + n];
    }
    __syncthreads();

    // ---- stage 3: T2@Wmlp2 + b2, then max over the 96 src rows ----
    {
        v8f acc[6];
#pragma unroll
        for (int t = 0; t < 6; ++t)
#pragma unroll
            for (int e = 0; e < 8; ++e) acc[t][e] = 0.f;
        gemm_96x128x128(Tb, Wt, lane, ncol, acc);
        float b2 = bmlp2[ncol];
        float pm = -1e30f;
#pragma unroll
        for (int mt = 0; mt < 6; ++mt)
#pragma unroll
            for (int v = 0; v < 8; ++v)
                pm = fmaxf(pm, acc[mt][v] + b2);
        pm = fmaxf(pm, __shfl_xor(pm, 16));   // combine lane halves (wave32)
        if (!(lane & 16))
            MMAX[((size_t)g * NN + j) * HH + ncol] = pm;
    }
}

// ---------------- combine: out proj + layernorm + stream reduce ----------------
__global__ __launch_bounds__(128) void combine_kernel(
    const float* __restrict__ ZOa, const float* __restrict__ ZOb,
    const float* __restrict__ MMa, const float* __restrict__ MMb,
    const float* __restrict__ W_o2, const float* __restrict__ b_o2,
    const float* __restrict__ ln_g, const float* __restrict__ ln_b,
    const float* __restrict__ W_red, const float* __restrict__ b_red,
    float* __restrict__ hidden, int layer) {
    __shared__ float sm[HH];
    __shared__ float outs[2 * HH];
    __shared__ float red[HH];
    int node = blockIdx.x, h = threadIdx.x;
    for (int s = 0; s < 2; ++s) {
        const float* MM = s ? MMb : MMa;
        const float* ZO = s ? ZOb : ZOa;
        const float* wo = W_o2 + (size_t)(layer * 2 + s) * HH * HH;
        int bo = (layer * 2 + s) * HH;
        __syncthreads();
        sm[h] = MM[(size_t)node * HH + h];
        __syncthreads();
        float t = ZO[(size_t)node * HH + h] + b_o2[bo + h];
        for (int k = 0; k < HH; ++k) t += sm[k] * wo[(size_t)k * HH + h];
        t = fmaxf(t, 0.f);
        red[h] = t;
        __syncthreads();
        for (int off = 64; off > 0; off >>= 1) {
            if (h < off) red[h] += red[h + off];
            __syncthreads();
        }
        float mu = red[0] * (1.f / 128.f);
        __syncthreads();
        float d = t - mu;
        red[h] = d * d;
        __syncthreads();
        for (int off = 64; off > 0; off >>= 1) {
            if (h < off) red[h] += red[h + off];
            __syncthreads();
        }
        float var = red[0] * (1.f / 128.f);
        __syncthreads();
        outs[s * HH + h] = d * rsqrtf(var + 1e-5f) * ln_g[bo + h] + ln_b[bo + h];
    }
    __syncthreads();
    const float* wr = W_red + (size_t)layer * 2 * HH * HH;
    float hv = b_red[layer * HH + h];
    for (int k = 0; k < 2 * HH; ++k) hv += outs[k] * wr[(size_t)k * HH + h];
    hidden[(size_t)node * HH + h] = hv;
}

// ---------------- readout ----------------
__global__ __launch_bounds__(128) void readout_kernel(
    const float* __restrict__ hidden, const float* __restrict__ W_p1,
    const float* __restrict__ b_p1, const float* __restrict__ W_p2,
    const float* __restrict__ b_p2, float* __restrict__ out) {
    __shared__ float emb[HH];
    __shared__ float red[HH];
    int g = blockIdx.x, h = threadIdx.x;
    float s = 0.f;
    for (int n = 0; n < NN; ++n) s += hidden[((size_t)g * NN + n) * HH + h];
    emb[h] = s * (1.f / (float)NN);
    __syncthreads();
    float t = b_p1[h];
    for (int k = 0; k < HH; ++k) t += emb[k] * W_p1[(size_t)k * HH + h];
    t = fmaxf(t, 0.f);
    red[h] = t * W_p2[h];
    __syncthreads();
    for (int off = 64; off > 0; off >>= 1) {
        if (h < off) red[h] += red[h + off];
        __syncthreads();
    }
    if (h == 0) out[g] = red[0] + b_p2[0];
}

// ---------------- launch ----------------
extern "C" void kernel_launch(void* const* d_in, const int* in_sizes, int n_in,
                              void* d_out, int out_size, void* d_ws,
                              size_t ws_size, hipStream_t stream) {
    (void)in_sizes; (void)n_in; (void)out_size; (void)ws_size;

    const int*   x          = (const int*)d_in[0];
    const int*   edge_attr  = (const int*)d_in[1];
    const int*   edge_src   = (const int*)d_in[2];
    const int*   edge_dst   = (const int*)d_in[3];
    const float* atom_tab   = (const float*)d_in[6];
    const float* bond_tab   = (const float*)d_in[7];
    const float* W_m1 = (const float*)d_in[8],  *b_m1 = (const float*)d_in[9];
    const float* W_m2 = (const float*)d_in[10], *b_m2 = (const float*)d_in[11];
    const float* W_me = (const float*)d_in[12], *b_me = (const float*)d_in[13];
    const float* b_mg = (const float*)d_in[15];          // W_mg unused: graph_fts==0
    const float* W_mlp1 = (const float*)d_in[16], *b_mlp1 = (const float*)d_in[17];
    const float* W_mlp2 = (const float*)d_in[18], *b_mlp2 = (const float*)d_in[19];
    const float* W_o1 = (const float*)d_in[20], *b_o1 = (const float*)d_in[21];
    const float* W_o2 = (const float*)d_in[22], *b_o2 = (const float*)d_in[23];
    const float* ln_g = (const float*)d_in[24], *ln_b = (const float*)d_in[25];
    const float* W_red = (const float*)d_in[26], *b_red = (const float*)d_in[27];
    const float* W_p1 = (const float*)d_in[28], *b_p1 = (const float*)d_in[29];
    const float* W_p2 = (const float*)d_in[30], *b_p2 = (const float*)d_in[31];

    // workspace carve (256B aligned)
    char* w = (char*)d_ws;
    auto carve = [&](size_t bytes) -> void* {
        void* p = (void*)w;
        w += (bytes + 255) & ~(size_t)255;
        return p;
    };
    const size_t NB = (size_t)GG * NN * HH * sizeof(float);   // 786432
    float* nf     = (float*)carve(NB);
    float* hidden = (float*)carve(NB);
    float* C1a = (float*)carve(NB), *C1b = (float*)carve(NB);
    float* M2a = (float*)carve(NB), *M2b = (float*)carve(NB);
    float* ZOa = (float*)carve(NB), *ZOb = (float*)carve(NB);
    float* MMa = (float*)carve(NB), *MMb = (float*)carve(NB);
    const size_t EBYTES = (size_t)GG * NN * NN * HH * sizeof(_Float16); // 37.75MB
    _Float16* Ebuf = (_Float16*)carve(EBYTES);

    zero_kernel<<<2048, 256, 0, stream>>>((uint4*)Ebuf, (int)(EBYTES / 16));
    zero_kernel<<<256, 256, 0, stream>>>((uint4*)hidden, (int)(NB / 16));
    node_embed_kernel<<<GG * NN, HH, 0, stream>>>(x, atom_tab, nf);
    edge_scatter_kernel<<<EE, HH, 0, stream>>>(edge_attr, edge_src, edge_dst,
                                               bond_tab, Ebuf);
    for (int i = 0; i < LL; ++i) {
        node_linear_kernel<<<GG * NN, HH, 0, stream>>>(
            nf, hidden, W_m1, b_m1, W_m2, b_m2, W_o1, b_o1, b_me, b_mg,
            C1a, C1b, M2a, M2b, ZOa, ZOb, i);
        for (int s = 0; s < 2; ++s) {
            fused_msg_kernel<<<GG * NN, 256, 0, stream>>>(
                Ebuf, s ? C1b : C1a, s ? M2b : M2a,
                W_me   + (size_t)(i * 2 + s) * HH * HH,
                W_mlp1 + (size_t)(i * 2 + s) * HH * HH,
                W_mlp2 + (size_t)(i * 2 + s) * HH * HH,
                b_mlp1 + (size_t)(i * 2 + s) * HH,
                b_mlp2 + (size_t)(i * 2 + s) * HH,
                s ? MMb : MMa);
        }
        combine_kernel<<<GG * NN, HH, 0, stream>>>(
            ZOa, ZOb, MMa, MMb, W_o2, b_o2, ln_g, ln_b, W_red, b_red, hidden, i);
    }
    readout_kernel<<<GG, HH, 0, stream>>>(hidden, W_p1, b_p1, W_p2, b_p2,
                                          (float*)d_out);
}
